// Net_7335804141892
// MI455X (gfx1250) — matrix operand order repaired and checked by reading, not compiled, not run on hardware
//
#include <hip/hip_runtime.h>
#include <hip/hip_bf16.h>
#include <math.h>

typedef __attribute__((ext_vector_type(16))) _Float16 v16h;
typedef __attribute__((ext_vector_type(8)))  _Float16 v8h;
typedef __attribute__((ext_vector_type(4)))  _Float16 v4h;
typedef __attribute__((ext_vector_type(8)))  float    v8f;

// ---------------------------------------------------------------------------
// Pack activations: f32 (N x Fin) -> f16 (N x FinPad), relu fused, zero-padded
// along K so the GEMM needs no K guards. FinPad is a power of two (shift).
// ---------------------------------------------------------------------------
__global__ __launch_bounds__(256)
void pack_a_f16(const float* __restrict__ in, _Float16* __restrict__ out,
                int Fin, int padShift, int reluIn, int total)
{
    int idx = blockIdx.x * blockDim.x + threadIdx.x;
    if (idx >= total) return;
    int n = idx >> padShift;
    int k = idx & ((1 << padShift) - 1);
    float v = (k < Fin) ? in[(long long)n * Fin + k] : 0.0f;
    if (reluIn) v = fmaxf(v, 0.0f);
    out[idx] = (_Float16)v;
}

// ---------------------------------------------------------------------------
// Pack weights transposed: f32 (Fin x Fout) -> f16 (Fout x FinPad), zero-pad K.
// Transposition makes each lane's B fragment K-contiguous in memory.
// ---------------------------------------------------------------------------
__global__ __launch_bounds__(256)
void pack_wT_f16(const float* __restrict__ w, _Float16* __restrict__ out,
                 int Fin, int Fout, int padShift, int total)
{
    int idx = blockIdx.x * blockDim.x + threadIdx.x;
    if (idx >= total) return;
    int col = idx >> padShift;
    int k   = idx & ((1 << padShift) - 1);
    out[idx] = (_Float16)((k < Fin) ? w[k * Fout + col] : 0.0f);
}

// ---------------------------------------------------------------------------
// Fused dual GEMM (WMMA f16 -> f32):
//   h_out(f16) = aP @ w1 ; acc_out(f32) = aP @ w2 + bias
// One wave32 per 16x16 tile of both outputs (shared A fragment).
// All inner-loop loads are unconditional 16-byte vector loads.
// A frag layout: k = kb + hs*16 + e               -> one contiguous run of 16.
// B frag layout: k = kb + hs*16 + e               -> one contiguous run of 16.
// ---------------------------------------------------------------------------
__global__ __launch_bounds__(32)
void gcs_gemm_wmma(const _Float16* __restrict__ aP,   // N x FinPad
                   const _Float16* __restrict__ w1T,  // Fout x FinPad
                   const _Float16* __restrict__ w2T,  // Fout x FinPad
                   const float*    __restrict__ bias,
                   _Float16* __restrict__ h_out,      // N x Fout (f16)
                   float*    __restrict__ acc_out,    // N x Fout (f32)
                   int Nn, int FinPad, int Fout)
{
    const int lane = threadIdx.x & 31;
    const int m    = lane & 15;   // A row-in-tile; B/C/D col-in-tile
    const int hs   = lane >> 4;   // half-select

    const int tileRow = blockIdx.x;
    const int tileCol = blockIdx.y;
    const int row     = tileRow * 16 + m;
    const int rowc    = (row < Nn) ? row : (Nn - 1);   // clamp: loads stay in range
    const int col     = tileCol * 16 + m;

    const _Float16* aRow  = aP  + (size_t)rowc * FinPad;
    const _Float16* b1Col = w1T + (size_t)col  * FinPad;
    const _Float16* b2Col = w2T + (size_t)col  * FinPad;

    v8f c1 = {};
    v8f c2 = {};

    for (int kb = 0; kb < FinPad; kb += 32) {
        v8h al = *(const v8h*)(aRow + kb + hs * 16);
        v8h ah = *(const v8h*)(aRow + kb + hs * 16 + 8);
        v16h a = __builtin_shufflevector(al, ah, 0,1,2,3,4,5,6,7,8,9,10,11,12,13,14,15);

        v8h p0 = *(const v8h*)(b1Col + kb + hs * 16);
        v8h p1 = *(const v8h*)(b1Col + kb + hs * 16 + 8);
        v16h b1 = __builtin_shufflevector(p0, p1, 0,1,2,3,4,5,6,7,8,9,10,11,12,13,14,15);

        v8h q0 = *(const v8h*)(b2Col + kb + hs * 16);
        v8h q1 = *(const v8h*)(b2Col + kb + hs * 16 + 8);
        v16h b2 = __builtin_shufflevector(q0, q1, 0,1,2,3,4,5,6,7,8,9,10,11,12,13,14,15);

        // (neg_a, A, neg_b, B, c_mod, C, reuse_a, reuse_b)
        c1 = __builtin_amdgcn_wmma_f32_16x16x32_f16(false, a, false, b1, (short)0, c1, false, false);
        c2 = __builtin_amdgcn_wmma_f32_16x16x32_f16(false, a, false, b2, (short)0, c2, false, false);
    }

    const float bv   = bias[col];
    const bool  full = (tileRow * 16 + 16) <= Nn;
#pragma unroll
    for (int r = 0; r < 8; ++r) {
        int orow = tileRow * 16 + r + 8 * hs;   // C/D layout: VGPR r -> M = r + 8*hs
        if (full || orow < Nn) {
            h_out  [(size_t)orow * Fout + col] = (_Float16)c1[r];
            acc_out[(size_t)orow * Fout + col] = c2[r] + bv;
        }
    }
}

// ---------------------------------------------------------------------------
// Edge scatter: acc[er[e], :] += ew[e] * h[ec[e], :]
// One thread per (edge, 4-feature chunk). h is f16 (halves gather bytes);
// accumulation stays f32 via global_atomic_add_f32 at L2.
// ---------------------------------------------------------------------------
__global__ __launch_bounds__(256)
void gcs_scatter(const _Float16* __restrict__ h, const float* __restrict__ ew,
                 const int* __restrict__ er, const int* __restrict__ ec,
                 float* __restrict__ acc, int Fout, int chunkShift, long long total)
{
    long long idx = (long long)blockIdx.x * blockDim.x + threadIdx.x;
    if (idx >= total) return;
    int e = (int)(idx >> chunkShift);
    int c = ((int)idx & ((1 << chunkShift) - 1)) << 2;   // feature offset (x4)

    int   src = ec[e];
    int   dst = er[e];
    float w   = ew[e];

    v4h v = *(const v4h*)(h + (size_t)src * Fout + c);
    float* out = acc + (size_t)dst * Fout + c;
    atomicAdd(out + 0, w * (float)v[0]);
    atomicAdd(out + 1, w * (float)v[1]);
    atomicAdd(out + 2, w * (float)v[2]);
    atomicAdd(out + 3, w * (float)v[3]);
}

// ---------------------------------------------------------------------------
// Zero helper (workspace is poisoned; pool accumulators need zeros per call).
// ---------------------------------------------------------------------------
__global__ void zero_f32(float* __restrict__ p, int n)
{
    int i = blockIdx.x * blockDim.x + threadIdx.x;
    if (i < n) p[i] = 0.0f;
}

// ---------------------------------------------------------------------------
// Global average pool numerators/counts, relu fused on read.
// ---------------------------------------------------------------------------
__global__ __launch_bounds__(256)
void gcs_pool(const float* __restrict__ acc, const int* __restrict__ seg,
              float* __restrict__ sums, float* __restrict__ cnts, int Nn)
{
    int idx = blockIdx.x * blockDim.x + threadIdx.x;
    if (idx >= Nn * 32) return;
    int n = idx >> 5;
    int f = idx & 31;
    float v = fmaxf(acc[(long long)n * 32 + f], 0.0f);
    int g = seg[n];
    atomicAdd(&sums[g * 32 + f], v);
    if (f == 0) atomicAdd(&cnts[g], 1.0f);
}

// ---------------------------------------------------------------------------
// mean -> 32x2 dense -> softmax. One thread per graph.
// ---------------------------------------------------------------------------
__global__ __launch_bounds__(256)
void gcs_head(const float* __restrict__ sums, const float* __restrict__ cnts,
              const float* __restrict__ wd, const float* __restrict__ bd,
              float* __restrict__ out, int G, int C)
{
    int g = blockIdx.x * blockDim.x + threadIdx.x;
    if (g >= G) return;
    float inv_cnt = 1.0f / fmaxf(cnts[g], 1.0f);
    float l0 = bd[0], l1 = bd[1];
#pragma unroll 8
    for (int f = 0; f < C; ++f) {
        float p = sums[g * C + f] * inv_cnt;
        l0 += p * wd[f * 2 + 0];
        l1 += p * wd[f * 2 + 1];
    }
    float mx = fmaxf(l0, l1);
    float e0 = __expf(l0 - mx);
    float e1 = __expf(l1 - mx);
    float inv = 1.0f / (e0 + e1);
    out[g * 2 + 0] = e0 * inv;
    out[g * 2 + 1] = e1 * inv;
}

extern "C" void kernel_launch(void* const* d_in, const int* in_sizes, int n_in,
                              void* d_out, int out_size, void* d_ws, size_t ws_size,
                              hipStream_t stream)
{
    const float* x    = (const float*)d_in[0];
    const float* ew   = (const float*)d_in[1];
    const int*   er   = (const int*)  d_in[2];
    const int*   ec   = (const int*)  d_in[3];
    const int*   seg  = (const int*)  d_in[4];
    const float* W1[3] = {(const float*)d_in[5], (const float*)d_in[8],  (const float*)d_in[11]};
    const float* W2[3] = {(const float*)d_in[6], (const float*)d_in[9],  (const float*)d_in[12]};
    const float* B [3] = {(const float*)d_in[7], (const float*)d_in[10], (const float*)d_in[13]};
    const float* wd   = (const float*)d_in[14];
    const float* bd   = (const float*)d_in[15];

    const int E  = in_sizes[1];          // 800000
    const int Nn = in_sizes[4];          // 50000
    const int F  = in_sizes[0] / Nn;     // 30
    const int G  = out_size / 2;         // 256

    // ---- Workspace layout ----
    float*    bufA   = (float*)d_ws;                         // N x 128 f32 (accum ping)
    float*    bufB   = bufA + (size_t)Nn * 128;              // N x 128 f32 (accum pong)
    float*    sums   = bufB + (size_t)Nn * 128;              // G x 32
    float*    cnts   = sums + (size_t)G * 32;                // G
    _Float16* bufH   = (_Float16*)(cnts + G);                // N x 128 f16 (h = xW1)
    _Float16* aPack  = bufH + (size_t)Nn * 128;              // N x 128 f16 (packed A)
    _Float16* w1T    = aPack + (size_t)Nn * 128;             // <=128x128 f16
    _Float16* w2T    = w1T + 16384;

    const int rowTiles = (Nn + 15) / 16;
    const dim3 wblk(32);

    struct { int Fin, FinPad, padShift, Fout; const float* in; float* acc; int relu; } L[3] = {
        { F,   32,  5, 128, x,    bufA, 0 },
        { 128, 128, 7, 64,  bufA, bufB, 1 },
        { 64,  64,  6, 32,  bufB, bufA, 1 },
    };

    for (int l = 0; l < 3; ++l) {
        const int Fin = L[l].Fin, FinPad = L[l].FinPad, sh = L[l].padShift, Fout = L[l].Fout;

        // Stage operands as padded f16 (relu fused for layers 2,3)
        int ta = Nn * FinPad;
        pack_a_f16<<<(ta + 255) / 256, 256, 0, stream>>>(L[l].in, aPack, Fin, sh, L[l].relu, ta);
        int tw = Fout * FinPad;
        pack_wT_f16<<<(tw + 255) / 256, 256, 0, stream>>>(W1[l], w1T, Fin, Fout, sh, tw);
        pack_wT_f16<<<(tw + 255) / 256, 256, 0, stream>>>(W2[l], w2T, Fin, Fout, sh, tw);

        // h = aP @ w1 (f16 out) ; acc = aP @ w2 + b (f32 out)
        gcs_gemm_wmma<<<dim3(rowTiles, Fout / 16), wblk, 0, stream>>>(
            aPack, w1T, w2T, B[l], bufH, L[l].acc, Nn, FinPad, Fout);

        // acc += scatter(ew * h[ec]) into rows er
        long long total = (long long)E * (Fout / 4);
        int chunkShift = (Fout == 128) ? 5 : (Fout == 64 ? 4 : 3);
        unsigned nb = (unsigned)((total + 255) / 256);
        gcs_scatter<<<dim3(nb), 256, 0, stream>>>(bufH, ew, er, ec, L[l].acc, Fout, chunkShift, total);
    }

    // ---- Pool (relu fused) + head ----
    zero_f32<<<(G * 33 + 255) / 256, 256, 0, stream>>>(sums, G * 33); // sums+cnts contiguous
    gcs_pool<<<(Nn * 32 + 255) / 256, 256, 0, stream>>>(bufA, seg, sums, cnts, Nn);
    gcs_head<<<(G + 255) / 256, 256, 0, stream>>>(sums, cnts, wd, bd, (float*)d_out, G, 32);
}
